// BasicBlock_27152783246056
// MI455X (gfx1250) — 8-bit matrix operand order repaired and checked by reading, compiled, not run on hardware
//
#include <hip/hip_runtime.h>
#include <stdint.h>

typedef __attribute__((ext_vector_type(8))) int v8i;
typedef int v4i_ __attribute__((vector_size(16)));

#define NB    64
#define CCH   256
#define NPIX  65536
#define KTOT  2304
#define KSTEPS 36

#define WS_XQ     0L
#define WS_A2Q    16777216L
#define WS_W1Q    33554432L
#define WS_W2Q    (WS_W1Q + 589824L)
#define WS_Y      (WS_W2Q + 589824L)
#define WS_PSUM   (WS_Y + 67108864L)
#define WS_PSUMSQ (WS_PSUM + 262144L)
#define WS_SCAL   (WS_PSUMSQ + 262144L)

struct Scal {
  unsigned amax_x, amax_w1, amax_w2;
  float scale_x, scale_w1, scale_w2;
  float s_conv1, act_scale, s_conv2;
  unsigned maxk[256], mink[256];
  float aArr[256], bArr[256], aArr2[256], bArr2[256];
};

__device__ __forceinline__ unsigned fkey(float f) {
  unsigned b = __float_as_uint(f);
  return (b & 0x80000000u) ? ~b : (b | 0x80000000u);
}
__device__ __forceinline__ float unkey(unsigned k) {
  unsigned b = (k & 0x80000000u) ? (k ^ 0x80000000u) : ~k;
  return __uint_as_float(b);
}

#if defined(__has_builtin)
# if __has_builtin(__builtin_amdgcn_global_load_async_to_lds_b128)
#  define HAVE_ASYNC_LDS 1
# endif
#endif

__device__ __forceinline__ void stage_b128(const int8_t* g, int8_t* l) {
#ifdef HAVE_ASYNC_LDS
  __builtin_amdgcn_global_load_async_to_lds_b128(
      (__attribute__((address_space(1))) v4i_*)(size_t)g,
      (__attribute__((address_space(3))) v4i_*)(unsigned)(size_t)l, 0, 0);
#else
  *(int4*)l = *(const int4*)g;
#endif
}

__device__ __forceinline__ void stage_wait() {
#ifdef HAVE_ASYNC_LDS
# if __has_builtin(__builtin_amdgcn_s_wait_asynccnt)
  __builtin_amdgcn_s_wait_asynccnt(0);
# else
  asm volatile("s_wait_asynccnt 0x0" ::: "memory");
# endif
#endif
}

__device__ __forceinline__ int swz(int row, int k) {
  return (row << 6) | (((k >> 4) ^ ((row >> 2) & 3)) << 4) | (k & 15);
}

__global__ void k_init(Scal* sc) {
  int t = threadIdx.x;
  if (t < 3) (&sc->amax_x)[t] = 0u;
  sc->maxk[t] = 0u;
  sc->mink[t] = 0xFFFFFFFFu;
}

__global__ void k_zero_stats(Scal* sc) {
  int t = threadIdx.x;
  sc->maxk[t] = 0u;
  sc->mink[t] = 0xFFFFFFFFu;
}

__global__ void k_absmax(const float* __restrict__ x, long n, unsigned* out) {
  __shared__ unsigned red[256];
  long i = (long)blockIdx.x * blockDim.x + threadIdx.x;
  long stride = (long)gridDim.x * blockDim.x;
  float m = 0.f;
  for (; i < n; i += stride) m = fmaxf(m, fabsf(x[i]));
  red[threadIdx.x] = __float_as_uint(m);
  __syncthreads();
  for (int s = 128; s > 0; s >>= 1) {
    if (threadIdx.x < s) red[threadIdx.x] = max(red[threadIdx.x], red[threadIdx.x + s]);
    __syncthreads();
  }
  if (threadIdx.x == 0) atomicMax(out, red[0]);
}

__global__ void k_scales(Scal* sc) {
  sc->scale_x  = __uint_as_float(sc->amax_x)  / 127.f + 1e-12f;
  sc->scale_w1 = __uint_as_float(sc->amax_w1) / 127.f + 1e-12f;
  sc->scale_w2 = __uint_as_float(sc->amax_w2) / 127.f + 1e-12f;
  sc->s_conv1  = sc->scale_x * sc->scale_w1;
}

__global__ void k_quant_x(const float* __restrict__ x, const Scal* __restrict__ sc,
                          int8_t* __restrict__ xq) {
  float inv = 1.f / sc->scale_x;
  int idx = blockIdx.x * 256 + threadIdx.x;
  int c4  = (idx & 63) << 2;
  long pix = idx >> 6;
  long n = pix >> 10, hw = pix & 1023;
  const float* src = x + n * 262144 + (long)c4 * 1024 + hw;
  char4 q;
  float v0 = fminf(fmaxf(rintf(src[0]    * inv), -128.f), 127.f);
  float v1 = fminf(fmaxf(rintf(src[1024] * inv), -128.f), 127.f);
  float v2 = fminf(fmaxf(rintf(src[2048] * inv), -128.f), 127.f);
  float v3 = fminf(fmaxf(rintf(src[3072] * inv), -128.f), 127.f);
  q.x = (signed char)(int)v0; q.y = (signed char)(int)v1;
  q.z = (signed char)(int)v2; q.w = (signed char)(int)v3;
  *(char4*)(xq + pix * 256 + c4) = q;
}

__global__ void k_quant_w(const float* __restrict__ w, const float* __restrict__ scalePtr,
                          int8_t* __restrict__ wq) {
  int o = blockIdx.x;
  float inv = 1.f / *scalePtr;
  for (int it = 0; it < 9; ++it) {
    int k  = it * 256 + threadIdx.x;
    int seg = k >> 8, ci = k & 255;
    int kh = seg / 3, kw = seg - kh * 3;
    float v = w[(long)o * 2304 + (long)ci * 9 + kh * 3 + kw];
    float q = fminf(fmaxf(rintf(v * inv), -128.f), 127.f);
    wq[(long)o * 2304 + k] = (int8_t)(int)q;
  }
}

template <bool SGN_A>
__device__ __forceinline__ void conv_stage(const int8_t* __restrict__ Aq,
                                           const int8_t* __restrict__ Wq,
                                           int8_t* __restrict__ AsBuf,
                                           int8_t* __restrict__ BsBuf,
                                           int pixBase, int chBase, int k0, int tid) {
  const int seg = k0 >> 8;
  const int dh = seg / 3 - 1, dw = seg % 3 - 1;
  const int c0 = k0 & 255;
#pragma unroll
  for (int cc = 0; cc < 2; ++cc) {
    const int ch  = tid + (cc << 8);
    const int row = ch >> 2, ckk = ch & 3;
    const int ldsOff = (row << 6) | (((ckk ^ ((row >> 2) & 3)) << 4));
    const int pix = pixBase + row;
    const int n = pix >> 10, h = (pix >> 5) & 31, w = pix & 31;
    const int hs = h + dh, wsd = w + dw;
    if ((unsigned)hs < 32u && (unsigned)wsd < 32u) {
      stage_b128(Aq + ((((long)(n << 10) + (hs << 5) + wsd)) << 8) + c0 + (ckk << 4),
                 AsBuf + ldsOff);
    } else {
      *(int4*)(AsBuf + ldsOff) = make_int4(0, 0, 0, 0);
    }
    stage_b128(Wq + (long)(chBase + row) * 2304 + k0 + (ckk << 4), BsBuf + ldsOff);
  }
}

template <bool SGN_A>
__global__ __launch_bounds__(256) void k_conv(const int8_t* __restrict__ Aq,
                                              const int8_t* __restrict__ Wq,
                                              float* __restrict__ Y,
                                              const float* __restrict__ sPtr) {
  __shared__ __align__(16) int8_t As[2][8192];
  __shared__ __align__(16) int8_t Bs[2][8192];
  const int tid  = threadIdx.x;
  const int pixBase = blockIdx.y * 128;
  const int chBase  = blockIdx.x * 128;
  const int wv = tid >> 5, lane = tid & 31;
  const int wm = wv & 3, wn = wv >> 2;
  const int l16 = lane & 15, hi = lane >> 4;
  const float s = *sPtr;

  v8i acc[2][4];
  const v8i vzero = {0, 0, 0, 0, 0, 0, 0, 0};
#pragma unroll
  for (int i = 0; i < 2; ++i)
#pragma unroll
    for (int j = 0; j < 4; ++j) acc[i][j] = vzero;

  conv_stage<SGN_A>(Aq, Wq, As[0], Bs[0], pixBase, chBase, 0, tid);
  stage_wait();
  __syncthreads();

  for (int ks = 0; ks < KSTEPS; ++ks) {
    const int cur = ks & 1;
    if (ks + 1 < KSTEPS)
      conv_stage<SGN_A>(Aq, Wq, As[cur ^ 1], Bs[cur ^ 1],
                        pixBase, chBase, (ks + 1) << 6, tid);

    const int8_t* Ab = As[cur];
    const int8_t* Bb = Bs[cur];
    v8i afrag[2], bfrag[4];
#pragma unroll
    for (int mf = 0; mf < 2; ++mf) {
      const int row = wm * 32 + mf * 16 + l16;
      const int* p0 = (const int*)(Ab + swz(row, hi * 16));
      const int* p1 = (const int*)(Ab + swz(row, 32 + hi * 16));
      afrag[mf][0] = p0[0]; afrag[mf][1] = p0[1];
      afrag[mf][2] = p0[2]; afrag[mf][3] = p0[3];
      afrag[mf][4] = p1[0]; afrag[mf][5] = p1[1];
      afrag[mf][6] = p1[2]; afrag[mf][7] = p1[3];
    }
#pragma unroll
    for (int nf = 0; nf < 4; ++nf) {
      const int col = wn * 64 + nf * 16 + l16;
      int4 b0 = *(const int4*)(Bb + swz(col, hi * 16));
      int4 b1 = *(const int4*)(Bb + swz(col, 32 + hi * 16));
      bfrag[nf][0] = b0.x; bfrag[nf][1] = b0.y; bfrag[nf][2] = b0.z; bfrag[nf][3] = b0.w;
      bfrag[nf][4] = b1.x; bfrag[nf][5] = b1.y; bfrag[nf][6] = b1.z; bfrag[nf][7] = b1.w;
    }
#pragma unroll
    for (int mf = 0; mf < 2; ++mf)
#pragma unroll
      for (int nf = 0; nf < 4; ++nf)
        acc[mf][nf] = __builtin_amdgcn_wmma_i32_16x16x64_iu8(
            SGN_A, afrag[mf], true, bfrag[nf], acc[mf][nf], false, false);

    stage_wait();
    __syncthreads();
  }

#pragma unroll
  for (int mf = 0; mf < 2; ++mf)
#pragma unroll
    for (int nf = 0; nf < 4; ++nf) {
      const int col  = chBase + wn * 64 + nf * 16 + l16;
      const int mrow = wm * 32 + mf * 16 + hi * 8;
#pragma unroll
      for (int v = 0; v < 8; ++v) {
        const long pix = pixBase + mrow + v;
        Y[pix * 256 + col] = s * (float)acc[mf][nf][v];
      }
    }
}

__global__ void k_bn_stats(const float* __restrict__ y, float* __restrict__ psum,
                           float* __restrict__ psumsq, Scal* sc) {
  const int b = blockIdx.x, t = threadIdx.x;
  const float* p = y + (long)b * 65536 + t;
  float s = 0.f, ss = 0.f, mx = -3.402823e38f, mn = 3.402823e38f;
  for (int r = 0; r < 256; ++r) {
    float v = p[r * 256];
    s += v; ss += v * v;
    mx = fmaxf(mx, v); mn = fminf(mn, v);
  }
  psum[b * 256 + t] = s;
  psumsq[b * 256 + t] = ss;
  atomicMax(&sc->maxk[t], fkey(mx));
  atomicMin(&sc->mink[t], fkey(mn));
}

__global__ void k_bn_finalize(const float* __restrict__ psum, const float* __restrict__ psumsq,
                              const float* __restrict__ gamma, const float* __restrict__ beta,
                              Scal* sc, float* __restrict__ aOut, float* __restrict__ bOut,
                              int computeAct) {
  __shared__ float red[256];
  const int t = threadIdx.x;
  float s = 0.f, ss = 0.f;
  for (int bk = 0; bk < 256; ++bk) {
    s  += psum[bk * 256 + t];
    ss += psumsq[bk * 256 + t];
  }
  const float NT = 65536.f;
  float mean = s / NT;
  float var  = ss / NT - mean * mean;
  float r = rsqrtf(var + 1e-5f);
  float a  = gamma[t] * r;
  float bb = beta[t] - mean * a;
  aOut[t] = a; bOut[t] = bb;
  if (computeAct) {
    float vmax = a * unkey(sc->maxk[t]) + bb;
    float vmin = a * unkey(sc->mink[t]) + bb;
    float m = fmaxf(0.f, fmaxf(vmax, vmin));
    red[t] = m; __syncthreads();
    for (int sdn = 128; sdn > 0; sdn >>= 1) {
      if (t < sdn) red[t] = fmaxf(red[t], red[t + sdn]);
      __syncthreads();
    }
    if (t == 0) {
      sc->act_scale = red[0] / 255.f + 1e-12f;
      sc->s_conv2 = sc->act_scale * sc->scale_w2;
    }
  }
}

__global__ void k_bnrq(const float* __restrict__ y, const Scal* __restrict__ sc,
                       uint8_t* __restrict__ aq) {
  int idx = blockIdx.x * 256 + threadIdx.x;
  int c = (idx & 63) << 2;
  long pix = idx >> 6;
  float inv = 1.f / sc->act_scale;
  float4 v = *(const float4*)(y + pix * 256 + c);
  uchar4 q;
  float u0 = fmaxf(0.f, sc->aArr[c + 0] * v.x + sc->bArr[c + 0]);
  float u1 = fmaxf(0.f, sc->aArr[c + 1] * v.y + sc->bArr[c + 1]);
  float u2 = fmaxf(0.f, sc->aArr[c + 2] * v.z + sc->bArr[c + 2]);
  float u3 = fmaxf(0.f, sc->aArr[c + 3] * v.w + sc->bArr[c + 3]);
  q.x = (unsigned char)(int)fminf(rintf(u0 * inv), 255.f);
  q.y = (unsigned char)(int)fminf(rintf(u1 * inv), 255.f);
  q.z = (unsigned char)(int)fminf(rintf(u2 * inv), 255.f);
  q.w = (unsigned char)(int)fminf(rintf(u3 * inv), 255.f);
  *(uchar4*)(aq + pix * 256 + c) = q;
}

__global__ void k_final(const float* __restrict__ y, const float* __restrict__ x,
                        const Scal* __restrict__ sc, float* __restrict__ out) {
  long idx = (long)blockIdx.x * 256 + threadIdx.x;
  int hw = idx & 1023;
  int c  = (int)((idx >> 10) & 255);
  long n = idx >> 18;
  long pix = (n << 10) + hw;
  float v = sc->aArr2[c] * y[pix * 256 + c] + sc->bArr2[c] + x[idx];
  out[idx] = fmaxf(v, 0.f);
}

extern "C" void kernel_launch(void* const* d_in, const int* in_sizes, int n_in,
                              void* d_out, int out_size, void* d_ws, size_t ws_size,
                              hipStream_t stream) {
  const float* x  = (const float*)d_in[0];
  const float* w1 = (const float*)d_in[1];
  const float* g1 = (const float*)d_in[2];
  const float* b1 = (const float*)d_in[3];
  const float* w2 = (const float*)d_in[4];
  const float* g2 = (const float*)d_in[5];
  const float* b2 = (const float*)d_in[6];
  char* ws = (char*)d_ws;

  int8_t*  xq   = (int8_t*)(ws + WS_XQ);
  uint8_t* a2q  = (uint8_t*)(ws + WS_A2Q);
  int8_t*  w1q  = (int8_t*)(ws + WS_W1Q);
  int8_t*  w2q  = (int8_t*)(ws + WS_W2Q);
  float*   y    = (float*)(ws + WS_Y);
  float*   psum = (float*)(ws + WS_PSUM);
  float*   psq  = (float*)(ws + WS_PSUMSQ);
  Scal*    sc   = (Scal*)(ws + WS_SCAL);

  k_init<<<1, 256, 0, stream>>>(sc);
  k_absmax<<<2048, 256, 0, stream>>>(x, 16777216L, &sc->amax_x);
  k_absmax<<<64, 256, 0, stream>>>(w1, 589824L, &sc->amax_w1);
  k_absmax<<<64, 256, 0, stream>>>(w2, 589824L, &sc->amax_w2);
  k_scales<<<1, 1, 0, stream>>>(sc);

  k_quant_x<<<16384, 256, 0, stream>>>(x, sc, xq);
  k_quant_w<<<256, 256, 0, stream>>>(w1, &sc->scale_w1, w1q);
  k_quant_w<<<256, 256, 0, stream>>>(w2, &sc->scale_w2, w2q);

  k_conv<true><<<dim3(2, 512), 256, 0, stream>>>(xq, w1q, y, &sc->s_conv1);
  k_bn_stats<<<256, 256, 0, stream>>>(y, psum, psq, sc);
  k_bn_finalize<<<1, 256, 0, stream>>>(psum, psq, g1, b1, sc, sc->aArr, sc->bArr, 1);
  k_bnrq<<<16384, 256, 0, stream>>>(y, sc, a2q);

  k_zero_stats<<<1, 256, 0, stream>>>(sc);
  k_conv<false><<<dim3(2, 512), 256, 0, stream>>>((const int8_t*)a2q, w2q, y, &sc->s_conv2);
  k_bn_stats<<<256, 256, 0, stream>>>(y, psum, psq, sc);
  k_bn_finalize<<<1, 256, 0, stream>>>(psum, psq, g2, b2, sc, sc->aArr2, sc->bArr2, 0);

  k_final<<<65536, 256, 0, stream>>>(y, x, sc, (float*)d_out);
}
